// FeedForwardQuantum_65481071395240
// MI455X (gfx1250) — compile-verified, matrix operand order checked by reading, not run on hardware
//
#include <hip/hip_runtime.h>

// ---- MI455X (gfx1250) fused quantum-FFN ----
// out[m,n] = b2[n] + sum_k relu(b1[k] + qout[m,:] . W1[k,:]) * W2[n,k]
// M=32768, N=512, K=2048, NQ=8. Compute-bound -> f16 WMMA, f32 accumulate.
// Per WG (256 thr / 8 waves): 64 rows x 512 cols. h stripe lives in LDS (256KB).
// Each wave: 64 rows x 64 cols (4x4 WMMA tiles), stages its own W2 panel with
// global_load_async_to_lds_b128 (double-buffered, wave-local sync via ASYNCcnt).

typedef __attribute__((ext_vector_type(16))) _Float16 v16h;
typedef __attribute__((ext_vector_type(8)))  _Float16 h8;
typedef __attribute__((ext_vector_type(8)))  float    v8f;
typedef __attribute__((ext_vector_type(4)))  float    f4;
typedef __attribute__((ext_vector_type(4)))  _Float16 h4;

#define M_TOTAL   32768
#define N_E       512
#define K_FFN     2048
#define NQQ       8
#define ROWS_WG   64
#define KC        32

// ---------------- Kernel 0: W2 f32 -> f16 into workspace ----------------
__global__ __launch_bounds__(256)
void cvt_w2_f16(const float* __restrict__ W2, _Float16* __restrict__ W2h) {
    int i = (blockIdx.x * 256 + threadIdx.x) * 4;
    f4 v = *(const f4*)(W2 + i);
    *(h4*)(W2h + i) = __builtin_convertvector(v, h4);
}

// Flat shared-pointer low 32 bits == LDS byte address (ISA aperture rule).
__device__ __forceinline__ unsigned lds_addr32(const void* p) {
    return (unsigned)(unsigned long long)p;
}
__device__ __forceinline__ void async_copy_b128(const void* gsrc, void* ldst) {
    asm volatile("global_load_async_to_lds_b128 %0, %1, off"
                 :: "v"(lds_addr32(ldst)), "v"((unsigned long long)gsrc)
                 : "memory");
}
__device__ __forceinline__ void wait_async0() {
    asm volatile("s_wait_asynccnt 0x0" ::: "memory");
}
__device__ __forceinline__ void wait_ds0() {
    asm volatile("s_wait_dscnt 0x0" ::: "memory");
}

// Wave stages its own 64-row x 32-k W2 panel (4KB) into LDS: 8 x b128 per lane.
__device__ __forceinline__ void stage_panel(const _Float16* __restrict__ W2h,
                                            _Float16* __restrict__ w2buf,
                                            int wave, int lane, int kc) {
#pragma unroll
    for (int i = 0; i < 8; ++i) {
        const int idx = lane * 8 + i;      // 0..255 chunks of 16B
        const int r   = idx >> 2;          // 0..63 row within panel
        const int kv  = idx & 3;           // 0..3  (8 f16 each)
        const int n   = wave * 64 + r;     // global output column
        async_copy_b128(W2h + (size_t)n * K_FFN + kc + kv * 8,
                        w2buf + n * KC + kv * 8);
    }
}

// ---------------- Kernel 1: fused ----------------
// LDS: h_sh [64][2048] f16 = 256KB | w2 double buffer 2 x [512][32] f16 = 64KB -> 320KB
__global__ __launch_bounds__(256)
void ffn_fused(const float* __restrict__ x,
               const float* __restrict__ theta,
               const float* __restrict__ W1,
               const float* __restrict__ b1,
               const _Float16* __restrict__ W2h,
               const float* __restrict__ b2,
               float* __restrict__ out) {
    extern __shared__ char smem[];
    _Float16* h_sh   = (_Float16*)smem;                                  // [64][2048]
    _Float16* w2_sh0 = (_Float16*)(smem + (size_t)ROWS_WG * K_FFN * 2);  // [512][32]
    _Float16* w2_sh1 = w2_sh0 + (size_t)N_E * KC;                        // [512][32]

    const int tid     = threadIdx.x;
    const int wave    = tid >> 5;
    const int lane    = tid & 31;
    const int wg_row0 = blockIdx.x * ROWS_WG;

    // Kick off async stage of first W2 panel; latency hides behind phase 1.
    stage_panel(W2h, w2_sh0, wave, lane, 0);

    // ---- Phase 1: h stripe [64 x 2048] -> LDS (GEMM1 + ReLU, computed once) ----
    // 8 k-results packed per ds_store_b128; W1/b1 reads merge into b128 loads.
    {
        const int row    = tid >> 2;               // 0..63
        const int kquart = tid & 3;                // 0..3
        const int gRow   = wg_row0 + row;
        float q[NQQ];
#pragma unroll
        for (int j = 0; j < NQQ; ++j)
            q[j] = __cosf(x[(size_t)gRow * N_E + j]) * __cosf(theta[j]);

        const int k0 = kquart * (K_FFN / 4);
        for (int k = k0; k < k0 + K_FFN / 4; k += 8) {
            float acc[8];
#pragma unroll
            for (int i = 0; i < 8; ++i) acc[i] = b1[k + i];
#pragma unroll
            for (int i = 0; i < 8; ++i) {
                const float* w1r = W1 + (size_t)(k + i) * NQQ;
#pragma unroll
                for (int j = 0; j < NQQ; ++j) acc[i] = fmaf(q[j], w1r[j], acc[i]);
            }
            h8 o;
#pragma unroll
            for (int i = 0; i < 8; ++i) o[i] = (_Float16)fmaxf(acc[i], 0.0f);
            *(h8*)(h_sh + row * K_FFN + k) = o;    // 16B LDS store
        }
    }
    __syncthreads();   // h_sh is read-only from here on; last barrier in the kernel

    // ---- Phase 2: GEMM2, 4x4 WMMA tiles per wave, wave-local async pipeline ----
    v8f c[4][4];
    {
        v8f z = {0.f,0.f,0.f,0.f,0.f,0.f,0.f,0.f};
#pragma unroll
        for (int mt = 0; mt < 4; ++mt)
#pragma unroll
            for (int nt = 0; nt < 4; ++nt) c[mt][nt] = z;
    }

    // A/B fragments use contiguous k (same order for both): lanes 0-15 carry K 0-15, lanes 16-31 carry K 16-31
    const int frow  = lane & 15;
    const int koff  = (lane >> 4) * 16;
    const int nbase = wave * 64;

    for (int kc = 0; kc < K_FFN; kc += KC) {
        _Float16* cur = ((kc >> 5) & 1) ? w2_sh1 : w2_sh0;
        _Float16* nxt = ((kc >> 5) & 1) ? w2_sh0 : w2_sh1;

        wait_async0();   // current panel resident in LDS (wave-local)
        wait_ds0();      // prior LDS reads of 'nxt' region fully retired
        if (kc + KC < K_FFN)
            stage_panel(W2h, nxt, wave, lane, kc + KC);  // overlaps with 16 WMMAs

        v16h a[4], b[4];
#pragma unroll
        for (int mt = 0; mt < 4; ++mt)
            a[mt] = *(const v16h*)(h_sh + (mt * 16 + frow) * K_FFN + kc + koff);
#pragma unroll
        for (int nt = 0; nt < 4; ++nt)
            b[nt] = *(const v16h*)(cur + (nbase + nt * 16 + frow) * KC + koff);

#pragma unroll
        for (int mt = 0; mt < 4; ++mt)
#pragma unroll
            for (int nt = 0; nt < 4; ++nt)
                c[mt][nt] = __builtin_amdgcn_wmma_f32_16x16x32_f16(
                                false, a[mt], false, b[nt],
                                (short)0, c[mt][nt], false, false);
    }

    // ---- Epilogue: C/D layout -> VGPR r: M=r (lanes 0-15) / M=8+r (lanes 16-31), N=lane&15
    const int crow = (lane >> 4) * 8;
    const int ccol = lane & 15;
#pragma unroll
    for (int mt = 0; mt < 4; ++mt)
#pragma unroll
        for (int nt = 0; nt < 4; ++nt) {
            const int n  = nbase + nt * 16 + ccol;
            const float bb = b2[n];
#pragma unroll
            for (int r = 0; r < 8; ++r) {
                const int m = wg_row0 + mt * 16 + crow + r;
                out[(size_t)m * N_E + n] = c[mt][nt][r] + bb;
            }
        }
}

// ---------------- Launcher ----------------
extern "C" void kernel_launch(void* const* d_in, const int* in_sizes, int n_in,
                              void* d_out, int out_size, void* d_ws, size_t ws_size,
                              hipStream_t stream) {
    const float* x     = (const float*)d_in[0];
    const float* theta = (const float*)d_in[1];
    const float* W1    = (const float*)d_in[2];
    const float* b1    = (const float*)d_in[3];
    const float* W2    = (const float*)d_in[4];
    const float* b2    = (const float*)d_in[5];
    float* out         = (float*)d_out;

    _Float16* W2h = (_Float16*)d_ws;   // 2MB f16 copy of W2

    cvt_w2_f16<<<(N_E * K_FFN / 4) / 256, 256, 0, stream>>>(W2, W2h);

    const size_t lds_bytes = (size_t)ROWS_WG * K_FFN * 2      // h stripe     256KB
                           + (size_t)2 * N_E * KC * 2;        // W2 dbl buf    64KB
    ffn_fused<<<M_TOTAL / ROWS_WG, 256, lds_bytes, stream>>>(x, theta, W1, b1, W2h, b2, out);
}
